// ZCANormBatch_4715874091903
// MI455X (gfx1250) — compile-verified, matrix operand order checked by reading, not run on hardware
//
#include <hip/hip_runtime.h>
#include <hip/hip_bf16.h>

// ---------------------------------------------------------------------------
// ZCA group whitening for x:(32,512,64,64), G=8 groups of 64 channels.
// Bandwidth-bound (~0.8 GB traffic -> ~35us floor at 23.3 TB/s).
// GEMMs: v_wmma_f32_16x16x32_bf16. Tile movement: TDM tensor_load_to_lds
// (TENSORcnt) with double buffering.
// ---------------------------------------------------------------------------

typedef __attribute__((ext_vector_type(16))) __bf16 v16bf;
typedef __attribute__((ext_vector_type(8)))  float  v8f;
typedef __attribute__((ext_vector_type(4)))  unsigned int v4u;
typedef __attribute__((ext_vector_type(8)))  int    v8i;
typedef __attribute__((ext_vector_type(4)))  int    v4i;

#define EPSC 0.01f
constexpr int G        = 8;
constexpr int NM       = 64;      // channels per group
constexpr int CCH      = 512;     // total channels
constexpr int HW       = 4096;    // H*W
constexpr int HW_MASK  = 4095;
constexpr int MTOT     = 131072;  // N*H*W per channel
constexpr int NCHUNK   = MTOT / 64;  // 64-wide m-chunks per group
constexpr int SPLIT1   = 64;      // blocks per group, covariance pass
constexpr int SPLIT3   = 128;     // blocks per group, apply pass

// [5/5] Pade of sqrt(1-x): from ((1+y)^11 +/- (1-y)^11), y^2 = 1-x.
__constant__ float PADE_P_C[6] = {1.0f, -2.75f, 2.75f, -1.203125f, 0.21484375f, -0.0107421875f};
__constant__ float PADE_Q_C[6] = {1.0f, -2.25f, 1.75f, -0.546875f, 0.05859375f, -0.0009765625f};

// ---------------------------------------------------------------------------
// TDM: DMA a 64x64 f32 tile (row stride HW elements) from global to LDS.
// D# per CDNA5 ISA 8.3/8.4. 2-D tensor -> groups 2/3 zero. Issued by one
// wave; completion via TENSORcnt. 6-arg builtin (clang-23 lane): the extra
// int32x8 group is passed zero-filled.
// ---------------------------------------------------------------------------
__device__ __forceinline__ void tdm_load_tile_64x64(unsigned lds_off, const float* gptr) {
  unsigned long long ga = (unsigned long long)(uintptr_t)gptr;
  v4u g0;
  g0[0] = 1u;                                            // count=1, user mode
  g0[1] = lds_off;                                       // lds_addr (bytes)
  g0[2] = (unsigned)(ga & 0xFFFFFFFFu);                  // global_addr[31:0]
  g0[3] = (unsigned)((ga >> 32) & 0x1FFFFFFu) | (2u << 30); // ga[56:32] | type=2
  const unsigned td0 = 1u << 20, td1 = 1u << 20;         // tensor dims (no OOB)
  const unsigned tile0 = 64u, tile1 = 64u;
  const unsigned long long st0 = (unsigned long long)HW; // dim0 stride (elems)
  v8i g1;
  g1[0] = (int)(2u << 16);                               // data_size=2 (4B), mask=0
  g1[1] = (int)((td0 & 0xFFFFu) << 16);                  // tensor_dim0[15:0]
  g1[2] = (int)(((td0 >> 16) & 0xFFFFu) | ((td1 & 0xFFFFu) << 16));
  g1[3] = (int)(((td1 >> 16) & 0xFFFFu) | (tile0 << 16));
  g1[4] = (int)tile1;                                    // tile_dim1 | tile_dim2=0
  g1[5] = (int)(st0 & 0xFFFFFFFFu);
  g1[6] = (int)((st0 >> 32) & 0xFFFFu);                  // | dim1_stride lo = 0
  g1[7] = 0;
  v4i gz4 = {0, 0, 0, 0};
  v8i gz8 = {0, 0, 0, 0, 0, 0, 0, 0};
  __builtin_amdgcn_tensor_load_to_lds(g0, g1, gz4, gz4, gz8, 0);
}

// K index inside a 16-bit A/B fragment (ISA 7.12.2, 16-bit A 16x32 layout).
__device__ __forceinline__ int k_of(int j, int half) {
  int v = j >> 1;
  return ((v >> 2) << 4) + (half << 3) + ((v & 3) << 1) + (j & 1);
}

// A (or mirrored B) fragment gathered from an f32 LDS tile, row-major.
__device__ __forceinline__ v16bf frag_row_f32(const float* t, int ld, int row, int kbase, int half) {
  v16bf r;
#pragma unroll
  for (int j = 0; j < 16; ++j) r[j] = (__bf16)t[row * ld + kbase + k_of(j, half)];
  return r;
}

// A fragment from a bf16 LDS tile, row-major.
__device__ __forceinline__ v16bf frag_row_bf16(const __bf16* t, int ld, int row, int kbase, int half) {
  v16bf r;
#pragma unroll
  for (int j = 0; j < 16; ++j) r[j] = t[row * ld + kbase + k_of(j, half)];
  return r;
}

// B fragment: B[k][n] = tile[k][col], tile row-major with leading dim ld.
__device__ __forceinline__ v16bf frag_col_f32(const float* t, int ld, int col, int kbase, int half) {
  v16bf r;
#pragma unroll
  for (int j = 0; j < 16; ++j) r[j] = (__bf16)t[(kbase + k_of(j, half)) * ld + col];
  return r;
}

__device__ __forceinline__ v8f wmma_bf16(v16bf a, v16bf b, v8f c) {
  return __builtin_amdgcn_wmma_f32_16x16x32_bf16(false, a, false, b, (short)0, c, false, false);
}

__device__ __forceinline__ size_t tile_goff(int g, int q) {
  const int n = q >> 6;                  // (q*64)/4096
  const int hw0 = (q << 6) & HW_MASK;
  return ((size_t)n * CCH + g * NM) * HW + hw0;
}

// ---------------------------------------------------------------------------
__global__ void zca_zero_kernel(float* __restrict__ p, int n) {
  int i = blockIdx.x * blockDim.x + threadIdx.x;
  if (i < n) p[i] = 0.0f;
}

// ---------------------------------------------------------------------------
// Pass 1: raw second moment S = X X^T and row sums. TDM double-buffered
// 64x64 tiles; 8 waves own the 16 16x16 output tiles (2 each, 4 WMMA/chunk).
// ---------------------------------------------------------------------------
__global__ __launch_bounds__(256) void zca_cov_kernel(const float* __restrict__ x,
                                                      float* __restrict__ cov,
                                                      float* __restrict__ sums) {
  __shared__ float sT[2][NM * 64];
  const int tid  = threadIdx.x;
  const int g    = blockIdx.x / SPLIT1;
  const int s    = blockIdx.x % SPLIT1;
  const int wave = tid >> 5;
  const int lane = tid & 31;
  const int half = lane >> 4;
  const int ln   = lane & 15;
  const int ti0  = wave >> 2;       // tiles (ti0,tj) and (ti0+2,tj)
  const int ti1  = ti0 + 2;
  const int tj   = wave & 3;
  const int srow = tid & 63;        // row-sum ownership
  const int squad = tid >> 6;

  v8f acc0 = {};
  v8f acc1 = {};
  float rowsum = 0.0f;

  if (wave == 0)                     // preload first chunk
    tdm_load_tile_64x64((unsigned)(uintptr_t)&sT[0][0], x + tile_goff(g, s));

  int buf = 0;
  for (int q = s; q < NCHUNK; q += SPLIT1) {
    if (wave == 0) {
      if (q + SPLIT1 < NCHUNK) {     // kick next DMA, then wait for current
        tdm_load_tile_64x64((unsigned)(uintptr_t)&sT[buf ^ 1][0],
                            x + tile_goff(g, q + SPLIT1));
        __builtin_amdgcn_s_wait_tensorcnt(1);
      } else {
        __builtin_amdgcn_s_wait_tensorcnt(0);
      }
    }
    __syncthreads();
    const float* t = &sT[buf][0];

#pragma unroll
    for (int k = 0; k < 16; ++k) rowsum += t[srow * 64 + squad * 16 + k];

    v16bf a0k0 = frag_row_f32(t, 64, ti0 * 16 + ln, 0,  half);
    v16bf a0k1 = frag_row_f32(t, 64, ti0 * 16 + ln, 32, half);
    v16bf a1k0 = frag_row_f32(t, 64, ti1 * 16 + ln, 0,  half);
    v16bf a1k1 = frag_row_f32(t, 64, ti1 * 16 + ln, 32, half);
    v16bf bk0  = frag_row_f32(t, 64, tj  * 16 + ln, 0,  half);  // B = X^T
    v16bf bk1  = frag_row_f32(t, 64, tj  * 16 + ln, 32, half);
    acc0 = wmma_bf16(a0k0, bk0, acc0);
    acc0 = wmma_bf16(a0k1, bk1, acc0);
    acc1 = wmma_bf16(a1k0, bk0, acc1);
    acc1 = wmma_bf16(a1k1, bk1, acc1);
    __syncthreads();
    buf ^= 1;
  }

  float* covg = cov + (size_t)g * NM * NM;
#pragma unroll
  for (int v = 0; v < 8; ++v) {                // C/D layout: M = v + 8*half
    int i0 = ti0 * 16 + v + 8 * half;
    int i1 = ti1 * 16 + v + 8 * half;
    int j  = tj * 16 + ln;
    atomicAdd(&covg[i0 * NM + j], acc0[v]);
    atomicAdd(&covg[i1 * NM + j], acc1[v]);
  }
  atomicAdd(&sums[g * NM + srow], rowsum);
}

// ---------------------------------------------------------------------------
// Matrix Horner: acc = coef[5]*I; acc = coef[k]*I + acc @ papp (k = 4..0)
// ---------------------------------------------------------------------------
__device__ void matpoly_horner(float* acc, float* tmp, const float* papp,
                               const float* coef, int tid) {
  for (int idx = tid; idx < NM * NM; idx += 256) {
    int i = idx >> 6, j = idx & 63;
    acc[idx] = (i == j) ? coef[5] : 0.0f;
  }
  __syncthreads();
  for (int k = 4; k >= 0; --k) {
    const int i  = tid >> 2;
    const int jb = (tid & 3) * 16;
    float r[16];
#pragma unroll
    for (int e = 0; e < 16; ++e) r[e] = 0.0f;
    for (int kk = 0; kk < NM; ++kk) {
      float a = acc[i * NM + kk];
#pragma unroll
      for (int e = 0; e < 16; ++e) r[e] += a * papp[kk * NM + jb + e];
    }
#pragma unroll
    for (int e = 0; e < 16; ++e) tmp[i * NM + jb + e] = r[e];
    __syncthreads();
    for (int idx = tid; idx < NM * NM; idx += 256) {
      int ii = idx >> 6, jj = idx & 63;
      acc[idx] = ((ii == jj) ? coef[k] : 0.0f) + tmp[idx];
    }
    __syncthreads();
  }
}

// ---------------------------------------------------------------------------
// Pass 2 (one block per group): finalize cov, Pade sqrt, solve psqrt*X=qsqrt,
// fold weight, beta = bias - W' mu. Store W' as bf16.
// ---------------------------------------------------------------------------
__global__ __launch_bounds__(256) void zca_whiten_kernel(const float* __restrict__ cov,
                                                         const float* __restrict__ sums,
                                                         const float* __restrict__ weight,
                                                         const float* __restrict__ bias,
                                                         float* __restrict__ psq_ws,
                                                         __bf16* __restrict__ Wb,
                                                         float* __restrict__ beta) {
  __shared__ float sA[NM * NM];
  __shared__ float sB[NM * NM];
  __shared__ float sC[NM * NM];
  __shared__ float sMu[NM];
  __shared__ float sRed[256];
  __shared__ float sNorm[1];

  const int tid = threadIdx.x;
  const int g   = blockIdx.x;
  const float invM = 1.0f / (float)MTOT;

  if (tid < NM) sMu[tid] = sums[g * NM + tid] * invM;
  __syncthreads();

  float ss = 0.0f;
  for (int idx = tid; idx < NM * NM; idx += 256) {
    int i = idx >> 6, j = idx & 63;
    float v = cov[(size_t)g * NM * NM + idx] * invM - sMu[i] * sMu[j] +
              ((i == j) ? EPSC : 0.0f);
    sA[idx] = v;
    ss += v * v;
  }
  sRed[tid] = ss;
  __syncthreads();
  for (int off = 128; off > 0; off >>= 1) {
    if (tid < off) sRed[tid] += sRed[tid + off];
    __syncthreads();
  }
  if (tid == 0) sNorm[0] = sqrtf(sRed[0]);
  __syncthreads();
  const float norm = sNorm[0];
  const float invN = 1.0f / norm;

  for (int idx = tid; idx < NM * NM; idx += 256) {   // sB = I - A/norm
    int i = idx >> 6, j = idx & 63;
    sB[idx] = ((i == j) ? 1.0f : 0.0f) - sA[idx] * invN;
  }
  __syncthreads();

  matpoly_horner(sC, sA, sB, PADE_P_C, tid);           // psqrt in sC
  for (int idx = tid; idx < NM * NM; idx += 256)       // stage psqrt (LDS budget)
    psq_ws[(size_t)g * NM * NM + idx] = sC[idx];
  __syncthreads();
  matpoly_horner(sC, sA, sB, PADE_Q_C, tid);           // qsqrt in sC
  for (int idx = tid; idx < NM * NM; idx += 256)
    sA[idx] = psq_ws[(size_t)g * NM * NM + idx];       // psqrt back to sA
  __syncthreads();

  // Gauss-Jordan: sA X = sC  ->  sC = inv(psqrt) @ qsqrt
  for (int k = 0; k < NM; ++k) {
    float pinv = 1.0f / sA[k * NM + k];
    __syncthreads();
    if (tid < 64)       sA[k * NM + tid] *= pinv;
    else if (tid < 128) sC[k * NM + (tid - 64)] *= pinv;
    __syncthreads();
    const int i   = tid & 63;
    const int seg = tid >> 6;
    float f = (i == k) ? 0.0f : sA[i * NM + k];
    __syncthreads();
    const int c0 = (seg & 1) * 32;
    if (seg < 2) {
#pragma unroll
      for (int c = 0; c < 32; ++c) sA[i * NM + c0 + c] -= f * sA[k * NM + c0 + c];
    } else {
#pragma unroll
      for (int c = 0; c < 32; ++c) sC[i * NM + c0 + c] -= f * sC[k * NM + c0 + c];
    }
    __syncthreads();
  }

  // W' = weight_row * X / sqrt(norm);  beta = bias - W' mu
  const float rs = rsqrtf(norm);
  {
    const int i = tid >> 2, part = tid & 3;
    const float wrow = weight[g * NM + i];
    float psum = 0.0f;
#pragma unroll
    for (int jj = 0; jj < 16; ++jj) {
      int j = part * 16 + jj;
      float w = sC[i * NM + j] * rs * wrow;
      Wb[(size_t)g * NM * NM + i * NM + j] = (__bf16)w;
      psum += w * sMu[j];
    }
    sRed[tid] = psum;
  }
  __syncthreads();
  if (tid < 64) {
    float bm = sRed[tid * 4] + sRed[tid * 4 + 1] + sRed[tid * 4 + 2] + sRed[tid * 4 + 3];
    beta[g * NM + tid] = bias[g * NM + tid] - bm;
  }
}

// ---------------------------------------------------------------------------
// Pass 3: out = W'(64x64) @ x_tile + beta. TDM double-buffered x tiles;
// W' fragments built once per block; 4 WMMAs per wave per chunk.
// ---------------------------------------------------------------------------
__global__ __launch_bounds__(256) void zca_apply_kernel(const float* __restrict__ x,
                                                        const __bf16* __restrict__ Wb,
                                                        const float* __restrict__ beta,
                                                        float* __restrict__ out) {
  __shared__ __bf16 sW[NM * NM];
  __shared__ float  sX[2][NM * 64];
  __shared__ float  sBeta[NM];

  const int tid  = threadIdx.x;
  const int g    = blockIdx.x / SPLIT3;
  const int s    = blockIdx.x % SPLIT3;
  const int wave = tid >> 5;
  const int lane = tid & 31;
  const int half = lane >> 4;
  const int ln   = lane & 15;
  const int ti0  = wave >> 2;
  const int ti1  = ti0 + 2;
  const int nt   = wave & 3;

  if (wave == 0)                     // preload first chunk via TDM
    tdm_load_tile_64x64((unsigned)(uintptr_t)&sX[0][0], x + tile_goff(g, s));

  for (int idx = tid; idx < NM * NM; idx += 256) sW[idx] = Wb[(size_t)g * NM * NM + idx];
  if (tid < NM) sBeta[tid] = beta[g * NM + tid];
  __syncthreads();

  const v16bf a00 = frag_row_bf16(sW, NM, ti0 * 16 + ln, 0,  half);
  const v16bf a01 = frag_row_bf16(sW, NM, ti0 * 16 + ln, 32, half);
  const v16bf a10 = frag_row_bf16(sW, NM, ti1 * 16 + ln, 0,  half);
  const v16bf a11 = frag_row_bf16(sW, NM, ti1 * 16 + ln, 32, half);

  int buf = 0;
  for (int q = s; q < NCHUNK; q += SPLIT3) {
    if (wave == 0) {
      if (q + SPLIT3 < NCHUNK) {
        tdm_load_tile_64x64((unsigned)(uintptr_t)&sX[buf ^ 1][0],
                            x + tile_goff(g, q + SPLIT3));
        __builtin_amdgcn_s_wait_tensorcnt(1);
      } else {
        __builtin_amdgcn_s_wait_tensorcnt(0);
      }
    }
    __syncthreads();
    const float* t = &sX[buf][0];

    v16bf b0 = frag_col_f32(t, 64, nt * 16 + ln, 0,  half);
    v16bf b1 = frag_col_f32(t, 64, nt * 16 + ln, 32, half);
    v8f acc0 = {};
    v8f acc1 = {};
    acc0 = wmma_bf16(a00, b0, acc0);
    acc0 = wmma_bf16(a01, b1, acc0);
    acc1 = wmma_bf16(a10, b0, acc1);
    acc1 = wmma_bf16(a11, b1, acc1);

    const size_t base = tile_goff(g, q);
#pragma unroll
    for (int v = 0; v < 8; ++v) {
      int r0 = ti0 * 16 + v + 8 * half;
      int r1 = ti1 * 16 + v + 8 * half;
      int mc = nt * 16 + ln;
      out[base + (size_t)r0 * HW + mc] = acc0[v] + sBeta[r0];
      out[base + (size_t)r1 * HW + mc] = acc1[v] + sBeta[r1];
    }
    __syncthreads();
    buf ^= 1;
  }
}

// ---------------------------------------------------------------------------
extern "C" void kernel_launch(void* const* d_in, const int* in_sizes, int n_in,
                              void* d_out, int out_size, void* d_ws, size_t ws_size,
                              hipStream_t stream) {
  (void)in_sizes; (void)n_in; (void)out_size; (void)ws_size;
  const float* x      = (const float*)d_in[0];
  const float* weight = (const float*)d_in[1];
  const float* bias   = (const float*)d_in[2];
  float* out = (float*)d_out;

  // workspace layout (~333 KB total)
  float*  cov  = (float*)d_ws;                 // G*64*64 f32 (raw X X^T sums)
  float*  sums = cov + G * NM * NM;            // G*64 f32   (row sums)
  float*  psq  = sums + G * NM;                // G*64*64 f32 (psqrt stage)
  __bf16* Wb   = (__bf16*)(psq + G * NM * NM); // G*64*64 bf16 (whitening mat)
  float*  beta = (float*)(Wb + G * NM * NM);   // G*64 f32

  const int nz = G * NM * NM + G * NM;         // cov + sums need zeroing
  zca_zero_kernel<<<(nz + 255) / 256, 256, 0, stream>>>(cov, nz);
  zca_cov_kernel<<<G * SPLIT1, 256, 0, stream>>>(x, cov, sums);
  zca_whiten_kernel<<<G, 256, 0, stream>>>(cov, sums, weight, bias, psq, Wb, beta);
  zca_apply_kernel<<<G * SPLIT3, 256, 0, stream>>>(x, Wb, beta, out);
}
